// RowLSTM_38689065402699
// MI455X (gfx1250) — compile-verified, matrix operand order checked by reading, not run on hardware
//
#include <hip/hip_runtime.h>
#include <hip/hip_bf16.h>

// ---------------------------------------------------------------------------
// RowLSTM fused kernel for MI455X (gfx1250, wave32, WMMA + TDM)
//
//   * conv(3,1) gates = GEMM M=256, K=192 (C*3 kh taps), N=w -> bf16 WMMA
//   * One block per (b,h): GEMM + bias + sigmoid/tanh + LSTM scan over W all
//     fused; activated gates live only in LDS (saves ~0.5 GB HBM round trip).
//   * x pre-converted to bf16 and transposed to [b][h][w][c] once, so each
//     block's three kh rows are contiguous 16KB blobs -> staged into LDS by
//     the Tensor Data Mover (tensor_load_to_lds + s_wait_tensorcnt).
//   * Activations: branch-free 2*sigmoid(2v)-1 trick with raw v_exp/v_rcp;
//     scan uses single-instruction v_tanh_f32.
//
//   LDS (dynamic; CDNA5 allows 320KB/WGP):
//     xs : staged x rows [kh=3][w=128][c=64] bf16 = 48 KB   (offset 0)
//     gl : activated gates [m=256][w=128]    bf16 = 64 KB
//     hb : h output buffer [hc=64][w=128]    f32  = 32 KB   total 144 KB
// ---------------------------------------------------------------------------

typedef __bf16 bf16_t;
typedef bf16_t v16bf __attribute__((ext_vector_type(16)));
typedef float  v8f   __attribute__((ext_vector_type(8)));
typedef unsigned int u32x4 __attribute__((ext_vector_type(4)));
typedef int          i32x4 __attribute__((ext_vector_type(4)));
typedef int          i32x8 __attribute__((ext_vector_type(8)));

#define B_   16
#define C_   64
#define H_   128
#define W_   128
#define M_   256     // 4 * Hc
#define K_   192     // C * 3
#define HC_  64
#define ROWB (W_ * C_)        // bf16 elements per (b,h) row = 8192

// ---- fast activations: raw gfx1250 transcendentals --------------------------
__device__ __forceinline__ float fast_sigmoid(float v) {
    // 1 / (1 + 2^(-v*log2(e)))
    const float e = __builtin_amdgcn_exp2f(-1.44269504088896340736f * v);
    return __builtin_amdgcn_rcpf(1.0f + e);
}
__device__ __forceinline__ float fast_tanh(float v) {
#if __has_builtin(__builtin_amdgcn_tanhf)
    return __builtin_amdgcn_tanhf(v);            // v_tanh_f32
#else
    const float e = __builtin_amdgcn_exp2f(2.88539008177792681472f * v);
    return 1.0f - 2.0f * __builtin_amdgcn_rcpf(e + 1.0f);
#endif
}

// ---------------------------------------------------------------------------
// Pack conv weights fp32 [M, C, 3, 1] -> bf16 [m][k], k = kh*64 + c.
// ---------------------------------------------------------------------------
__global__ void pack_weights_kernel(const float* __restrict__ w,
                                    bf16_t* __restrict__ wA) {
    int t = blockIdx.x * blockDim.x + threadIdx.x;
    if (t >= M_ * K_) return;
    int m  = t / K_;
    int k  = t % K_;
    int kh = k / C_;
    int c  = k % C_;
    wA[t] = (bf16_t)w[(m * C_ + c) * 3 + kh];
}

// ---------------------------------------------------------------------------
// Convert + transpose x fp32 [b][c][h][w] -> bf16 [b][h][w][c], so each
// (b,h) row is one contiguous 16KB blob in TDM/B-fragment order.
// ---------------------------------------------------------------------------
__global__ void pack_x_kernel(const float* __restrict__ x,
                              bf16_t* __restrict__ xt) {
    size_t t = (size_t)blockIdx.x * blockDim.x + threadIdx.x;   // 16M elems
    int w = (int)(t & (W_ - 1));
    size_t r = t >> 7;
    int hh = (int)(r & (H_ - 1)); r >>= 7;
    int c  = (int)(r & (C_ - 1));
    int b  = (int)(r >> 6);
    xt[(((size_t)b * H_ + hh) * W_ + w) * C_ + c] = (bf16_t)x[t];
}

// Per-VGPR starting K inside a 32-wide chunk (CDNA5 ISA 7.12.2, 16-bit A/B)
__device__ __forceinline__ int kpair(int j, int laneHi) {
    return (j < 4) ? (2 * j + (laneHi ? 8 : 0))
                   : (16 + 2 * (j - 4) + (laneHi ? 8 : 0));
}

__global__ void __launch_bounds__(512)
rowlstm_fused_kernel(const bf16_t* __restrict__ xbf,
                     const bf16_t* __restrict__ wA,
                     const float* __restrict__ bias,
                     float* __restrict__ out) {
    extern __shared__ char smem[];
    bf16_t* xs = (bf16_t*)smem;                                   // [3][W][C]
    bf16_t* gl = (bf16_t*)(smem + 3 * ROWB * 2);                  // [M][W]
    float*  hb = (float*)(smem + 3 * ROWB * 2 + M_ * W_ * 2);     // [HC][W]

    const int blk    = blockIdx.x;
    const int b      = blk / H_;
    const int h      = blk % H_;
    const int tid    = threadIdx.x;
    const int lane   = tid & 31;
    const int wv     = tid >> 5;        // wave id 0..15 == M-tile id
    const int laneHi = lane >> 4;
    const int ln16   = lane & 15;

    // ===== stage x rows h-2, h-1, h (each a contiguous 16KB bf16 blob) =====
#if __has_builtin(__builtin_amdgcn_tensor_load_to_lds)
    if (wv == 0) {
        // Wave 0 drives the Tensor Data Mover: one D# per kh row.
        #pragma unroll
        for (int r = 0; r < 3; ++r) {
            const int hh = h - 2 + r;
            if (hh >= 0) {
                const unsigned long long ga = (unsigned long long)(size_t)
                    (xbf + ((size_t)b * H_ + hh) * ROWB);
                const unsigned int lds_off = (unsigned int)(r * ROWB * 2);
                // ---- D# group 0 (ISA 8.3): count=1, lds_addr, global_addr, type=2
                u32x4 g0;
                g0[0] = 1u;                                   // count=1, user mode
                g0[1] = lds_off;                              // lds_addr [63:32]
                g0[2] = (unsigned int)(ga & 0xffffffffu);     // global_addr lo
                g0[3] = (unsigned int)((ga >> 32) & 0x1ffffffu) | (2u << 30); // hi|type=2
                // ---- D# group 1 (ISA 8.4): data_size=2B, one row of 8192 elems
                i32x8 g1;
                g1[0] = 0x10000;                 // wg_mask=0, data_size=1 (2 bytes)
                g1[1] = (ROWB & 0xffff) << 16;   // tensor_dim0[15:0] @bit48
                g1[2] = (1 << 16) | (ROWB >> 16);// tensor_dim0[31:16], tensor_dim1=1
                g1[3] = (ROWB & 0xffff) << 16;   // tile_dim0 @bit112
                g1[4] = 1;                       // tile_dim1=1
                g1[5] = ROWB;                    // tensor_dim0_stride lo
                g1[6] = 0;
                g1[7] = 0;
                const i32x4 z4 = {0, 0, 0, 0};
#if __clang_major__ >= 23
                const i32x8 z8 = {0, 0, 0, 0, 0, 0, 0, 0};
                __builtin_amdgcn_tensor_load_to_lds(g0, g1, z4, z4, z8, 0);
#else
                __builtin_amdgcn_tensor_load_to_lds(g0, g1, z4, z4, 0);
#endif
            }
        }
        __builtin_amdgcn_s_wait_tensorcnt(0);
    }
    // zero-fill causally-padded rows (hh < 0) -- disjoint from TDM targets
    if (h < 2) {
        for (int r = 0; r < 2 - h; ++r) {
            unsigned int* zp = (unsigned int*)(xs + r * ROWB);
            for (int z = tid; z < ROWB / 2; z += 512) zp[z] = 0u;
        }
    }
#else
    // Fallback: cooperative 16-byte copies
    for (int r = 0; r < 3; ++r) {
        const int hh = h - 2 + r;
        uint4* dst = (uint4*)(xs + r * ROWB);
        if (hh >= 0) {
            const uint4* src = (const uint4*)(xbf + ((size_t)b * H_ + hh) * ROWB);
            for (int q = tid; q < ROWB / 8; q += 512) dst[q] = src[q];
        } else {
            const uint4 zz = {0, 0, 0, 0};
            for (int q = tid; q < ROWB / 8; q += 512) dst[q] = zz;
        }
    }
#endif

    // ===== A fragments (weights): loaded once, held in VGPRs ================
    v16bf a[6];
    const int mA = wv * 16 + ln16;
    #pragma unroll
    for (int qc = 0; qc < 6; ++qc) {
        #pragma unroll
        for (int e = 0; e < 16; ++e) {
            const int kk = 32 * qc + kpair(e >> 1, laneHi) + (e & 1);
            a[qc][e] = wA[mA * K_ + kk];
        }
    }
    float bsv[8];
    #pragma unroll
    for (int r = 0; r < 8; ++r)
        bsv[r] = bias[wv * 16 + (laneHi ? 8 : 0) + r];

    // Branch-free activation select (wave-uniform):
    //   waves 0-11 (i/f/o): g = sigmoid(v)          -> sb = 1
    //   waves 12-15 (g):    g = tanh(v) = 2*sigmoid(2v) - 1 -> sb = 2
    const float sb   = (wv < 12) ? 1.0f : 2.0f;
    const float soff = sb - 1.0f;

    __syncthreads();

    // ===== GEMM: 8 N-tiles of 16 w-columns, K=192 in 6 chunks of 32 =========
    #pragma unroll 1
    for (int t = 0; t < 8; ++t) {
        v8f acc = {};
        const int wcol = t * 16 + ln16;
        #pragma unroll
        for (int qc = 0; qc < 6; ++qc) {
            const int kh   = qc >> 1;
            const int coff = (qc & 1) * 32;
            v16bf bf;
            #pragma unroll
            for (int e = 0; e < 16; ++e) {
                const int cc = coff + kpair(e >> 1, laneHi) + (e & 1);
                bf[e] = xs[(kh * W_ + wcol) * C_ + cc];
            }
            acc = __builtin_amdgcn_wmma_f32_16x16x32_bf16(
                      false, a[qc], false, bf, (short)0, acc, false, false);
        }
        // epilogue: bias + activation (no EXEC divergence), gates -> LDS bf16
        #pragma unroll
        for (int r = 0; r < 8; ++r) {
            const int m = wv * 16 + (laneHi ? 8 : 0) + r;
            const float v = acc[r] + bsv[r];
            const float g = __builtin_fmaf(sb, fast_sigmoid(sb * v), -soff);
            gl[m * W_ + wcol] = (bf16_t)g;
        }
    }

    __syncthreads();

    // ===== LSTM recurrence over W, entirely out of LDS ======================
    if (tid < HC_) {
        const int hc = tid;
        const bf16_t* gi = gl + (0 * HC_ + hc) * W_;
        const bf16_t* gf = gl + (1 * HC_ + hc) * W_;
        const bf16_t* go = gl + (2 * HC_ + hc) * W_;
        const bf16_t* gg = gl + (3 * HC_ + hc) * W_;
        float* hrow = hb + hc * W_;
        float c = 0.0f;
        #pragma unroll 4
        for (int w = 0; w < W_; ++w) {
            const float iv = (float)gi[w];
            const float fv = (float)gf[w];
            const float ov = (float)go[w];
            const float gv = (float)gg[w];
            c = fv * c + iv * gv;
            hrow[w] = ov * fast_tanh(c);
        }
    }

    __syncthreads();

    // ===== coalesced float4 store: out[b][hc][h][w] =========================
    for (int it = 0; it < 4; ++it) {
        int q  = it * 512 + tid;
        int hc = q >> 5;
        int w4 = (q & 31) * 4;
        const float4 v = *(const float4*)&hb[hc * W_ + w4];
        *(float4*)&out[(((size_t)b * HC_ + hc) * H_ + h) * W_ + w4] = v;
    }
}

// ---------------------------------------------------------------------------
extern "C" void kernel_launch(void* const* d_in, const int* in_sizes, int n_in,
                              void* d_out, int out_size, void* d_ws, size_t ws_size,
                              hipStream_t stream) {
    const float* x      = (const float*)d_in[0];   // [16,64,128,128]
    const float* conv_w = (const float*)d_in[1];   // [256,64,3,1]
    const float* conv_b = (const float*)d_in[2];   // [256]
    float* out = (float*)d_out;                    // [16,64,128,128]

    bf16_t* wA = (bf16_t*)d_ws;                               // 96 KB
    bf16_t* xt = (bf16_t*)((char*)d_ws + (128 << 10));        // 32 MB @ +128KB

    pack_weights_kernel<<<(M_ * K_ + 255) / 256, 256, 0, stream>>>(conv_w, wA);

    const size_t nx = (size_t)B_ * C_ * H_ * W_;
    pack_x_kernel<<<(unsigned)(nx / 256), 256, 0, stream>>>(x, xt);

    const size_t shmem = (size_t)3 * ROWB * 2     // xs  48 KB
                       + (size_t)M_ * W_ * 2      // gl  64 KB
                       + (size_t)HC_ * W_ * 4;    // hb  32 KB => 144 KB
    rowlstm_fused_kernel<<<B_ * H_, 512, shmem, stream>>>(xt, wA, conv_b, out);
}
